// NeighborCooccurrenceEncoder_24275155157491
// MI455X (gfx1250) — matrix operand order repaired and checked by reading, not compiled, not run on hardware
//
#include <hip/hip_runtime.h>

typedef _Float16 v8h  __attribute__((ext_vector_type(8)));
typedef _Float16 v16h __attribute__((ext_vector_type(16)));
typedef float    v8f  __attribute__((ext_vector_type(8)));

#define Bc 4096
#define Lc 128
#define Fc 64
#define NODE_MAXc 1024
#define APAD 72   // row stride in halves (64 + 8 pad), 144B = multiple of 16B

__global__ __launch_bounds__(256) void ncoe_fused_kernel(
    const long long* __restrict__ src_ids,
    const long long* __restrict__ dst_ids,
    const float* __restrict__ W1,
    const float* __restrict__ b1,
    const float* __restrict__ W2,
    const float* __restrict__ b2,
    float* __restrict__ out)
{
    // ---- LDS ----
    __shared__ unsigned hist[2][NODE_MAXc];                            // 8 KB
    __shared__ float sW1[Fc], sb1[Fc], sb2[Fc];                        // 768 B
    __shared__ __attribute__((aligned(16))) _Float16 W2t[Fc][APAD];    // B^T: [n][k], 9 KB
    __shared__ __attribute__((aligned(16))) _Float16 Ah[2*Lc][APAD];   // hsum rows, 36 KB

    const int tid = threadIdx.x;
    const int b   = blockIdx.x;

    // ---- stage weights + zero histograms ----
    for (int i = tid; i < 2*NODE_MAXc; i += 256) ((unsigned*)hist)[i] = 0u;
    if (tid < Fc)            sW1[tid]        = W1[tid];
    else if (tid < 2*Fc)     sb1[tid - Fc]   = b1[tid - Fc];
    else if (tid < 3*Fc)     sb2[tid - 2*Fc] = b2[tid - 2*Fc];
    for (int i = tid; i < Fc*Fc; i += 256) {              // W2[k][n] -> W2t[n][k], f16
        int k = i >> 6, n = i & 63;
        W2t[n][k] = (_Float16)W2[i];
    }
    __syncthreads();

    // ---- per-row histograms (threads 0..127: src row, 128..255: dst row) ----
    {
        const int side = tid >> 7;
        const int pos  = tid & 127;
        const long long* ids = side ? dst_ids : src_ids;
        const int id = (int)ids[(size_t)b * Lc + pos];
        atomicAdd(&hist[side][id], 1u);
    }
    __syncthreads();

    // ---- hsum = relu(f0*W1+b1) + relu(f1*W1+b1), one M-row per thread ----
    {
        const int r    = tid;            // 0..255 : [src rows | dst rows]
        const int side = r >> 7;
        const int pos  = r & 127;
        const long long* ids = side ? dst_ids : src_ids;
        const int id = (int)ids[(size_t)b * Lc + pos];
        float f0 = 0.f, f1 = 0.f;
        if (id != 0) {                    // mask zeroes freq, NOT the MLP output
            f0 = (float)hist[side][id];       // count in own row
            f1 = (float)hist[side ^ 1][id];   // count in partner row
        }
        #pragma unroll
        for (int k = 0; k < Fc; k += 8) {
            v8h pack;
            #pragma unroll
            for (int j = 0; j < 8; ++j) {
                const float w  = sW1[k + j];
                const float bb = sb1[k + j];
                const float h  = fmaxf(fmaf(f0, w, bb), 0.f)
                               + fmaxf(fmaf(f1, w, bb), 0.f);
                pack[j] = (_Float16)h;
            }
            *(v8h*)&Ah[r][k] = pack;          // ds_store_b128
        }
    }
    __syncthreads();

    // ---- GEMM: [256 x 64] x [64 x 64] via v_wmma_f32_16x16x32_f16 ----
    const int lane  = tid & 31;
    const int wave  = tid >> 5;
    const int lhalf = lane >> 4;     // lane half 0/1
    const int l15   = lane & 15;
    const size_t dst_base = (size_t)Bc * Lc * Fc;   // dst_feat offset in d_out

    #pragma unroll
    for (int t = 0; t < 8; ++t) {
        const int tile  = wave + t * 8;   // 0..63
        const int mtile = tile >> 2;      // 0..15
        const int ntile = tile & 3;       // 0..3
        const int row   = mtile * 16 + l15;    // A row this lane supplies
        const int coln  = ntile * 16 + l15;    // B column / D column this lane holds

        // C init: + 2*b2 (bias counted once per channel c in the reference sum)
        v8f c;
        const float cinit = 2.0f * sb2[coln];
        #pragma unroll
        for (int i = 0; i < 8; ++i) c[i] = cinit;

        #pragma unroll
        for (int kk = 0; kk < 2; ++kk) {
            // A 16x32 f16 frag: lane-half h holds K = {8h..8h+7, 16+8h..23+8h}
            const int abase = kk * 32 + lhalf * 8;
            v8h alo = *(const v8h*)&Ah[row][abase];
            v8h ahi = *(const v8h*)&Ah[row][abase + 16];
            v16h a = __builtin_shufflevector(alo, ahi,
                     0,1,2,3,4,5,6,7,8,9,10,11,12,13,14,15);
            // B 32x16 f16 frag: lane-half h holds K = {8h..8h+7, 16+8h..23+8h} of its column
            const int bbase = kk * 32 + lhalf * 8;
            v8h blo = *(const v8h*)&W2t[coln][bbase];
            v8h bhi = *(const v8h*)&W2t[coln][bbase + 16];
            v16h bf = __builtin_shufflevector(blo, bhi,
                      0,1,2,3,4,5,6,7,8,9,10,11,12,13,14,15);
            c = __builtin_amdgcn_wmma_f32_16x16x32_f16(
                    false, a, false, bf, (short)0, c, false, false);
        }

        // D layout: VGPR i -> M = mtile*16 + i + 8*lhalf, N = coln. Stream out NT.
        #pragma unroll
        for (int i = 0; i < 8; ++i) {
            const int m    = mtile * 16 + i + 8 * lhalf;   // 0..255
            const int side = m >> 7;
            const int pos  = m & 127;
            const size_t off = (side ? dst_base : 0)
                             + ((size_t)b * Lc + pos) * (size_t)Fc + coln;
            __builtin_nontemporal_store(c[i], &out[off]);
        }
    }
}

extern "C" void kernel_launch(void* const* d_in, const int* in_sizes, int n_in,
                              void* d_out, int out_size, void* d_ws, size_t ws_size,
                              hipStream_t stream) {
    const long long* src = (const long long*)d_in[0];  // int64 ids [B,L]
    const long long* dst = (const long long*)d_in[1];  // int64 ids [B,L]
    const float* W1 = (const float*)d_in[2];           // [1,F] -> 64
    const float* b1 = (const float*)d_in[3];           // [F]
    const float* W2 = (const float*)d_in[4];           // [F,F]
    const float* b2 = (const float*)d_in[5];           // [F]
    float* out = (float*)d_out;                        // src_feat | dst_feat, f32

    hipLaunchKernelGGL(ncoe_fused_kernel, dim3(Bc), dim3(256), 0, stream,
                       src, dst, W1, b1, W2, b2, out);
}
